// SelfAttention_33071248179891
// MI455X (gfx1250) — matrix operand order repaired and checked by reading, not compiled, not run on hardware
//
#include <hip/hip_runtime.h>
#include <hip/hip_bf16.h>

// ---------------------------------------------------------------------------
// Self-attention with 2D relative position bias, CDNA5 (gfx1250, wave32).
// B=8, H=W=32 (S=1024), D=256, heads=8, Dh=256, inner=2048.
// All matmuls via v_wmma_f32_16x16x32_bf16 (f32 accumulate); kernel-3 A-tile
// staging uses global_load_async_to_lds_b128 (ASYNCcnt path).
// ---------------------------------------------------------------------------

typedef __bf16 bf16;
typedef __attribute__((ext_vector_type(16))) __bf16 v16bf;
typedef __attribute__((ext_vector_type(8)))  __bf16 v8bf;
typedef __attribute__((ext_vector_type(4)))  __bf16 v4bf;
typedef __attribute__((ext_vector_type(8)))  float  v8f;

union V16U { v16bf v; v8bf h[2]; };

#define WMMA_BF16(a, b, c) \
  __builtin_amdgcn_wmma_f32_16x16x32_bf16(false, (a), false, (b), (short)0, (c), false, false)

static __device__ __forceinline__ v8f zero_v8f() {
  v8f z;
#pragma unroll
  for (int i = 0; i < 8; ++i) z[i] = 0.f;
  return z;
}

// A-matrix 16x32 bf16 fragment: lane holds row m = lane&15,
// one contiguous 16-element K run: k = hl*16..+15 (matches B).
static __device__ __forceinline__ v16bf load_a16x32(const bf16* rowbase, int hl) {
  V16U u;
  u.h[0] = *(const v8bf*)(rowbase + hl * 16);
  u.h[1] = *(const v8bf*)(rowbase + hl * 16 + 8);
  return u.v;
}
// B-matrix 32x16 fragment: lane holds column n = lane&15, contiguous K run of
// 16 at k = (lane>>4)*16 -> one 32-byte load when B is stored n-major.

// ===========================================================================
// Kernel 1: C[8192 x 2048] = X[8192 x 256] @ W[256 x 2048] + bias, stored as
// bf16 in attention layout. transposeOut=0: [b*8+h][s][d]; =1: [b*8+h][d][s].
// Block 256 thr (8 waves). BM=128, BN=128, BK=32. Waves 2(M)x4(N).
// ===========================================================================
__global__ __launch_bounds__(256) void qkv_gemm_kernel(
    const float* __restrict__ X, const float* __restrict__ W,
    const float* __restrict__ bias, bf16* __restrict__ out, int transposeOut)
{
  __shared__ __align__(64) bf16 As[128 * 40];   // [m][k], stride 40
  __shared__ __align__(64) bf16 Bt[128 * 48];   // [n][k], stride 48

  const int tid  = threadIdx.x;
  const int lane = tid & 31, hl = lane >> 4, ln = lane & 15;
  const int wav  = tid >> 5;
  const int rw   = wav >> 2;          // 0..1 : M offset rw*64
  const int cw   = wav & 3;           // 0..3 : N offset cw*32
  const int M0   = blockIdx.x * 128;
  const int N0   = blockIdx.y * 128;

  v8f acc[4][2];
#pragma unroll
  for (int mt = 0; mt < 4; ++mt)
#pragma unroll
    for (int nt = 0; nt < 2; ++nt) acc[mt][nt] = zero_v8f();

  for (int k0 = 0; k0 < 256; k0 += 32) {
    // ---- load phase: 8 b128 loads in flight (4 for A, 4 for B) ----
    float4 xa[4], wb[4];
#pragma unroll
    for (int it = 0; it < 4; ++it) {               // A: 128x32 in 4-float chunks
      int i = tid + it * 256;
      int m = i >> 3, kc = (i & 7) * 4;
      xa[it] = *(const float4*)&X[(size_t)(M0 + m) * 256 + k0 + kc];
    }
#pragma unroll
    for (int it = 0; it < 4; ++it) {               // B: 32x128 in 4-float chunks
      int i = tid + it * 256;
      int n4 = (i & 31) * 4, k = i >> 5;
      wb[it] = *(const float4*)&W[(size_t)(k0 + k) * 2048 + N0 + n4];
    }
    // ---- convert/store phase ----
#pragma unroll
    for (int it = 0; it < 4; ++it) {
      int i = tid + it * 256;
      int m = i >> 3, kc = (i & 7) * 4;
      v4bf pk = { (bf16)xa[it].x, (bf16)xa[it].y, (bf16)xa[it].z, (bf16)xa[it].w };
      *(v4bf*)&As[m * 40 + kc] = pk;
    }
#pragma unroll
    for (int it = 0; it < 4; ++it) {
      int i = tid + it * 256;
      int n4 = (i & 31) * 4, k = i >> 5;
      Bt[(n4 + 0) * 48 + k] = (bf16)wb[it].x;
      Bt[(n4 + 1) * 48 + k] = (bf16)wb[it].y;
      Bt[(n4 + 2) * 48 + k] = (bf16)wb[it].z;
      Bt[(n4 + 3) * 48 + k] = (bf16)wb[it].w;
    }
    __syncthreads();

    v16bf a[4], b[2];
#pragma unroll
    for (int mt = 0; mt < 4; ++mt)
      a[mt] = load_a16x32(&As[(rw * 64 + mt * 16 + ln) * 40], hl);
#pragma unroll
    for (int nt = 0; nt < 2; ++nt)
      b[nt] = *(const v16bf*)&Bt[(cw * 32 + nt * 16 + ln) * 48 + hl * 16];
#pragma unroll
    for (int mt = 0; mt < 4; ++mt)
#pragma unroll
      for (int nt = 0; nt < 2; ++nt)
        acc[mt][nt] = WMMA_BF16(a[mt], b[nt], acc[mt][nt]);
    __syncthreads();
  }

  // C/D layout: VGPR r -> row r + 8*hl, col ln (ISA 7.12.2).
#pragma unroll
  for (int mt = 0; mt < 4; ++mt)
#pragma unroll
    for (int nt = 0; nt < 2; ++nt)
#pragma unroll
      for (int r = 0; r < 8; ++r) {
        int Mg = M0 + rw * 64 + mt * 16 + r + 8 * hl;
        int Ng = N0 + cw * 32 + nt * 16 + ln;
        float v = acc[mt][nt][r] + bias[Ng];
        int bb = Mg >> 10, s = Mg & 1023;   // batch, seq
        int hh = Ng >> 8,  d = Ng & 255;    // head, dh
        size_t idx = transposeOut
                   ? ((size_t)(bb * 8 + hh) * 256 + d) * 1024 + s
                   : ((size_t)(bb * 8 + hh) * 1024 + s) * 256 + d;
        out[idx] = (bf16)v;
      }
}

// ===========================================================================
// Kernel 2: flash attention per (b,h). Block = 128 thr (4 waves), each wave
// owns 16 query rows, full Dh=256 accumulated in registers. 32-key tiles
// (one spatial row ki per tile -> rel-pos bias is rs[q][ki] + cs[q][kj]).
// ===========================================================================
__global__ __launch_bounds__(128) void flash_attn_kernel(
    const bf16* __restrict__ Q,   // [64][1024][256]
    const bf16* __restrict__ K,   // [64][1024][256]
    const bf16* __restrict__ Vt,  // [64][256][1024]  (transposed)
    const float* __restrict__ row_tab,  // [63][128]
    const float* __restrict__ col_tab,  // [63][128]
    bf16* __restrict__ attO)      // [8192][2048] row=b*1024+s, col=h*256+d
{
  __shared__ __align__(64) float rs[4][16][32];
  __shared__ __align__(64) float cs[4][16][32];
  __shared__ __align__(64) bf16  Pex[4][16][40];   // P tile, stride 40

  const int tid  = threadIdx.x;
  const int lane = tid & 31, hl = lane >> 4, ln = lane & 15;
  const int wav  = tid >> 5;
  const int bh   = blockIdx.y;           // 0..63
  const int b    = bh >> 3, h = bh & 7;
  const int qb   = blockIdx.x * 64 + wav * 16;

  const bf16* Qh = Q  + (size_t)bh * 1024 * 256;
  const bf16* Kh = K  + (size_t)bh * 1024 * 256;
  const bf16* Vh = Vt + (size_t)bh * 256 * 1024;

  // ---- relative position bias tables for this wave's 16 queries ----
  // rs[q][ki] = Q[sq, 0:128] . row_tab[ki-i+31];  cs[q][kj] = Q[sq,128:] . col_tab[j-kj+31]
#pragma unroll 1
  for (int it = 0; it < 16; ++it) {
    int p = lane + it * 32;
    int q = p >> 5, k = p & 31;
    int sq = qb + q;
    int i = sq >> 5, j = sq & 31;
    const bf16*  qr = Qh + (size_t)sq * 256;
    const float* rt = row_tab + (size_t)(k - i + 31) * 128;
    const float* ct = col_tab + (size_t)(j - k + 31) * 128;
    float a0 = 0.f, a1 = 0.f;
    for (int d = 0; d < 128; ++d) {
      a0 += (float)qr[d]       * rt[d];
      a1 += (float)qr[128 + d] * ct[d];
    }
    rs[wav][q][k] = a0;
    cs[wav][q][k] = a1;
  }
  __syncthreads();

  // ---- Q fragments held in registers: 8 k-chunks of 32 over Dh=256 ----
  v16bf qf[8];
  const bf16* qrow = Qh + (size_t)(qb + ln) * 256;
#pragma unroll
  for (int kk = 0; kk < 8; ++kk) qf[kk] = load_a16x32(qrow + kk * 32, hl);

  v8f o[16];
#pragma unroll
  for (int t = 0; t < 16; ++t) o[t] = zero_v8f();
  float mrow[8], lrow[8];
#pragma unroll
  for (int r = 0; r < 8; ++r) { mrow[r] = -1e30f; lrow[r] = 0.f; }

  for (int kt = 0; kt < 32; ++kt) {
    const int keyb = kt * 32;            // spatial row ki = kt for all 32 keys

    if (kt + 1 < 32) {                   // prefetch next K/V tiles into caches
      __builtin_prefetch(Kh + (size_t)(keyb + 32 + lane) * 256, 0, 0);
      __builtin_prefetch(Vh + (size_t)(lane * 8) * 1024 + keyb + 32, 0, 0);
    }

    v8f s0 = zero_v8f(), s1 = zero_v8f();
#pragma unroll
    for (int kk = 0; kk < 8; ++kk) {
      const bf16* kb = Kh + (size_t)(keyb + ln) * 256 + kk * 32 + hl * 16;
      v16bf b0 = *(const v16bf*)kb;                  // keys keyb..+15
      v16bf b1 = *(const v16bf*)(kb + 16 * 256);     // keys keyb+16..+31
      s0 = WMMA_BF16(qf[kk], b0, s0);
      s1 = WMMA_BF16(qf[kk], b1, s1);
    }

    // ---- bias + online softmax update (row = 16 lanes of one half) ----
    float a8[8];
#pragma unroll
    for (int r = 0; r < 8; ++r) {
      int ql = r + 8 * hl;
      float br = rs[wav][ql][kt];
      float e0 = s0[r] + br + cs[wav][ql][ln];
      float e1 = s1[r] + br + cs[wav][ql][16 + ln];
      float t = fmaxf(e0, e1);
#pragma unroll
      for (int off = 1; off < 16; off <<= 1) t = fmaxf(t, __shfl_xor(t, off, 16));
      float nm    = fmaxf(mrow[r], t);
      float alpha = __expf(mrow[r] - nm);
      mrow[r] = nm;
      float p0 = __expf(e0 - nm), p1 = __expf(e1 - nm);
      float ps = p0 + p1;
#pragma unroll
      for (int off = 1; off < 16; off <<= 1) ps += __shfl_xor(ps, off, 16);
      lrow[r] = lrow[r] * alpha + ps;
      a8[r] = alpha;
      Pex[wav][ql][ln]      = (bf16)p0;   // same-wave DS: in-order, no barrier
      Pex[wav][ql][16 + ln] = (bf16)p1;
    }
#pragma unroll
    for (int t = 0; t < 16; ++t)
#pragma unroll
      for (int r = 0; r < 8; ++r) o[t][r] *= a8[r];

    // ---- O += P(16x32) @ V(32x256): Vt gives contiguous B fragments ----
    v16bf pf = load_a16x32(&Pex[wav][ln][0], hl);
#pragma unroll
    for (int t = 0; t < 16; ++t) {
      v16bf vb = *(const v16bf*)(Vh + (size_t)(t * 16 + ln) * 1024 + keyb + hl * 16);
      o[t] = WMMA_BF16(pf, vb, o[t]);
    }
  }

  // ---- normalize and store attended (bf16) for the output projection ----
  float inv[8];
#pragma unroll
  for (int r = 0; r < 8; ++r) inv[r] = 1.f / lrow[r];
#pragma unroll
  for (int t = 0; t < 16; ++t)
#pragma unroll
    for (int r = 0; r < 8; ++r) {
      int sq  = qb + r + 8 * hl;
      int row = b * 1024 + sq;
      int col = h * 256 + t * 16 + ln;
      attO[(size_t)row * 2048 + col] = (bf16)(o[t][r] * inv[r]);
    }
}

// ===========================================================================
// Kernel 3: out[8192 x 256] = att[8192 x 2048](bf16) @ Pw[2048 x 256] + Pb.
// Block 256 thr (8 waves), BM=64, BN=64, BK=32; waves 4(M)x2(N).
// A-tile staged with global_load_async_to_lds_b128 (ASYNCcnt path).
// ===========================================================================
__global__ __launch_bounds__(256) void proj_gemm_kernel(
    const bf16* __restrict__ A, const float* __restrict__ W,
    const float* __restrict__ bias, float* __restrict__ out)
{
  __shared__ __align__(64) bf16 As[64 * 40];
  __shared__ __align__(64) bf16 Bt[64 * 48];

  const int tid  = threadIdx.x;
  const int lane = tid & 31, hl = lane >> 4, ln = lane & 15;
  const int wav  = tid >> 5;
  const int mw   = wav >> 1;            // 0..3 : m tile
  const int nw   = wav & 1;             // 0..1 : n offset nw*32
  const int M0   = blockIdx.x * 64, N0 = blockIdx.y * 64;

  // LDS byte offset for async dest = low 32 bits of the generic address
  // (ISA 10.2: LDS aperture addresses truncate to addr[31:0]).
  const int am = tid >> 2, akc = (tid & 3) * 8;          // 16B chunk per thread
  const unsigned lds_dst = (unsigned)(uintptr_t)(void*)&As[am * 40 + akc];

  v8f acc0 = zero_v8f(), acc1 = zero_v8f();

  for (int k0 = 0; k0 < 2048; k0 += 32) {
    // stage A tile (bf16 copy) via async DMA to LDS: 256 x 16B = 4KB
    {
      const bf16* gsrc = &A[(size_t)(M0 + am) * 2048 + k0 + akc];
      asm volatile("global_load_async_to_lds_b128 %0, %1, off"
                   :: "v"(lds_dst), "v"(gsrc) : "memory");
    }
    // stage B transposed: both float4 loads in flight, then convert/store
    float4 wb[2];
#pragma unroll
    for (int it = 0; it < 2; ++it) {
      int i = tid + it * 256;
      int n4 = (i & 15) * 4, k = i >> 4;
      wb[it] = *(const float4*)&W[(size_t)(k0 + k) * 256 + N0 + n4];
    }
#pragma unroll
    for (int it = 0; it < 2; ++it) {
      int i = tid + it * 256;
      int n4 = (i & 15) * 4, k = i >> 4;
      Bt[(n4 + 0) * 48 + k] = (bf16)wb[it].x;
      Bt[(n4 + 1) * 48 + k] = (bf16)wb[it].y;
      Bt[(n4 + 2) * 48 + k] = (bf16)wb[it].z;
      Bt[(n4 + 3) * 48 + k] = (bf16)wb[it].w;
    }
    asm volatile("s_wait_asynccnt 0x0" ::: "memory");
    __syncthreads();

    v16bf a  = load_a16x32(&As[(mw * 16 + ln) * 40], hl);
    v16bf b0 = *(const v16bf*)&Bt[(nw * 32 + ln) * 48 + hl * 16];
    v16bf b1 = *(const v16bf*)&Bt[(nw * 32 + 16 + ln) * 48 + hl * 16];
    acc0 = WMMA_BF16(a, b0, acc0);
    acc1 = WMMA_BF16(a, b1, acc1);
    __syncthreads();
  }

#pragma unroll
  for (int r = 0; r < 8; ++r) {
    int Mg = M0 + mw * 16 + r + 8 * hl;
    int Ng = N0 + nw * 32 + ln;
    out[(size_t)Mg * 256 + Ng]      = acc0[r] + bias[Ng];
    out[(size_t)Mg * 256 + Ng + 16] = acc1[r] + bias[Ng + 16];
  }
}

// ===========================================================================
extern "C" void kernel_launch(void* const* d_in, const int* in_sizes, int n_in,
                              void* d_out, int out_size, void* d_ws, size_t ws_size,
                              hipStream_t stream) {
  (void)in_sizes; (void)n_in; (void)out_size; (void)ws_size;
  const float* x   = (const float*)d_in[0];
  const float* Kw  = (const float*)d_in[1];
  const float* Kb_ = (const float*)d_in[2];
  const float* Qw  = (const float*)d_in[3];
  const float* Qb_ = (const float*)d_in[4];
  const float* Vw  = (const float*)d_in[5];
  const float* Vb_ = (const float*)d_in[6];
  const float* Pw  = (const float*)d_in[7];
  const float* Pb_ = (const float*)d_in[8];
  const float* row_tab = (const float*)d_in[9];
  const float* col_tab = (const float*)d_in[10];
  float* out = (float*)d_out;

  char* ws   = (char*)d_ws;                       // 4 x 32MB regions
  bf16* Qbf  = (bf16*)(ws);
  bf16* Kbf  = (bf16*)(ws + ((size_t)32 << 20));
  bf16* Vtr  = (bf16*)(ws + ((size_t)64 << 20));
  bf16* attO = (bf16*)(ws + ((size_t)96 << 20));

  dim3 g1(64, 16);
  qkv_gemm_kernel<<<g1, 256, 0, stream>>>(x, Qw, Qb_, Qbf, 0);
  qkv_gemm_kernel<<<g1, 256, 0, stream>>>(x, Kw, Kb_, Kbf, 0);
  qkv_gemm_kernel<<<g1, 256, 0, stream>>>(x, Vw, Vb_, Vtr, 1);

  flash_attn_kernel<<<dim3(16, 64), 128, 0, stream>>>(Qbf, Kbf, Vtr,
                                                      row_tab, col_tab, attO);

  proj_gemm_kernel<<<dim3(128, 4), 256, 0, stream>>>(attO, Pw, Pb_, out);
}
